// CustomGRULayer_11355893531348
// MI455X (gfx1250) — matrix operand order repaired and checked by reading, not compiled, not run on hardware
//
#include <hip/hip_runtime.h>

#define B_SZ 256
#define S_SZ 512
#define DIN  256
#define DH   512

typedef __attribute__((ext_vector_type(16))) __bf16 v16bf;
typedef __attribute__((ext_vector_type(8)))  float  v8f;

struct alignas(16) U4 { unsigned int x, y, z, w; };
struct alignas(16) F4 { float x, y, z, w; };

union BFrag {
  v16bf v;
  U4 q[2];
  unsigned short us[16];
};

__device__ __forceinline__ unsigned short f2bf(float f) {
  unsigned int u = __float_as_uint(f);
  u += 0x7FFFu + ((u >> 16) & 1u);          // round-to-nearest-even
  return (unsigned short)(u >> 16);
}

// Fast gate nonlinearities (v_exp_f32 + v_rcp_f32; saturate correctly at +-inf).
__device__ __forceinline__ float fast_sigmoid(float x) {
  return __builtin_amdgcn_rcpf(1.0f + __expf(-x));
}
__device__ __forceinline__ float fast_tanh(float x) {
  return 1.0f - 2.0f * __builtin_amdgcn_rcpf(1.0f + __expf(2.0f * x));
}

// Async global->LDS 16B copy (ASYNCcnt-tracked); fallback = plain copy.
#if defined(__gfx1250__)
__device__ __forceinline__ void async_cp16(void* lds_dst, const void* gsrc) {
  asm volatile("global_load_async_to_lds_b128 %0, %1, off"
               :: "v"((unsigned)(size_t)lds_dst), "v"(gsrc)
               : "memory");
}
#define WAIT_ASYNC() asm volatile("s_wait_asynccnt 0x0" ::: "memory")
#else
__device__ __forceinline__ void async_cp16(void* lds_dst, const void* gsrc) {
  *(U4*)lds_dst = *(const U4*)gsrc;
}
#define WAIT_ASYNC()
#endif

// A fragment (16x32 bf16, ISA 7.12.2): lane<16 -> M=lane, K runs {0..7,16..23};
// lane>=16 -> M=lane-16, K runs {8..15,24..31}. 'row' already includes M.
__device__ __forceinline__ v16bf load_a_frag(const unsigned short* row, int k0, int akoff) {
  BFrag f;
  const unsigned short* p = row + k0 + akoff;
  f.q[0] = *(const U4*)(p);
  f.q[1] = *(const U4*)(p + 16);
  return f.v;
}

// B fragment (32x16 bf16): N = lane&15; K runs {0..7,16..23} (lane<16) or {8..15,24..31}.
// Row-major W(K,N): W[(k0+K)*ldw + n0 + (lane&15)].
__device__ __forceinline__ v16bf load_b_w(const unsigned short* __restrict__ w,
                                          int ldw, int k0, int n0, int lane) {
  BFrag f;
  const unsigned short* p = w + (size_t)k0 * ldw + n0 + (lane & 15);
  const int kb = (lane >> 4) << 3;
#pragma unroll
  for (int s = 0; s < 16; ++s)
    f.us[s] = p[(size_t)(kb + (s & 7) + ((s >> 3) << 4)) * ldw];
  return f.v;
}

#define WMMA_BF16(A, B, C) \
  __builtin_amdgcn_wmma_f32_16x16x32_bf16(false, (A), false, (B), (short)0, (C), false, false)

// Stage A = [panel0 (16 x DH bf16), panelX (16 x KX)] into LDS (row stride LDA).
// XF32: panelX is fp32, converted while staging; else bf16 async-copied.
template <int KX, bool XF32>
__device__ __forceinline__ void stage_A(
    unsigned short* sA, int m0,
    const unsigned short* __restrict__ p0_bf,
    const unsigned short* __restrict__ xin_bf, const float* __restrict__ xin_f32,
    long xstride) {
  constexpr int LDA = DH + KX + 8;           // +16B row padding: bank spread
  const int tid = threadIdx.x;
#pragma unroll
  for (int c = tid; c < (16 * DH) >> 3; c += 256) {   // 4 full rounds
    const int row = c >> 6;
    const int col = (c & 63) << 3;
    async_cp16(&sA[row * LDA + col], p0_bf + (size_t)(m0 + row) * DH + col);
  }
  constexpr int CSH = (KX == 512) ? 6 : 5;   // 16B chunks per row
  if (XF32) {
#pragma unroll
    for (int c = tid; c < (16 * KX) >> 3; c += 256) {
      const int row = c >> CSH;
      const int col = (c & ((1 << CSH) - 1)) << 3;
      const float* p = xin_f32 + (size_t)(m0 + row) * xstride + col;
      F4 a0 = *(const F4*)p;
      F4 a1 = *(const F4*)(p + 4);
      U4 u;
      u.x = (unsigned)f2bf(a0.x) | ((unsigned)f2bf(a0.y) << 16);
      u.y = (unsigned)f2bf(a0.z) | ((unsigned)f2bf(a0.w) << 16);
      u.z = (unsigned)f2bf(a1.x) | ((unsigned)f2bf(a1.y) << 16);
      u.w = (unsigned)f2bf(a1.z) | ((unsigned)f2bf(a1.w) << 16);
      *(U4*)&sA[row * LDA + DH + col] = u;
    }
  } else {
#pragma unroll
    for (int c = tid; c < (16 * KX) >> 3; c += 256) {
      const int row = c >> CSH;
      const int col = (c & ((1 << CSH) - 1)) << 3;
      async_cp16(&sA[row * LDA + DH + col], xin_bf + (size_t)(m0 + row) * DH + col);
    }
  }
  WAIT_ASYNC();
  __syncthreads();
}

// One 16x16 C tile: A from LDS (ds_load_b128), B streamed from global
// (L2-resident weights). Dual accumulators, fully unrolled K for deep
// load/WMMA software pipelining.
template <int KTOT>
__device__ __forceinline__ v8f gemm_tile(
    const unsigned short* sA, const unsigned short* __restrict__ W, int ldw,
    int n0, int lane, int am, int akoff) {
  constexpr int LDA = KTOT + 8;
  v8f acc0 = {}, acc1 = {};
  const unsigned short* arow = sA + am * LDA;
#pragma unroll
  for (int k0 = 0; k0 < KTOT; k0 += 64) {
    v16bf a0 = load_a_frag(arow, k0,      akoff);
    v16bf b0 = load_b_w(W, ldw, k0,      n0, lane);
    v16bf a1 = load_a_frag(arow, k0 + 32, akoff);
    v16bf b1 = load_b_w(W, ldw, k0 + 32, n0, lane);
    acc0 = WMMA_BF16(a0, b0, acc0);
    acc1 = WMMA_BF16(a1, b1, acc1);
  }
  return acc0 + acc1;
}

// Gates: [h, x] @ Wz[:, :2H] + bz ; z=sigmoid(col<H), r=sigmoid(col>=H).
// Writes z (f32) and r*h (bf16). 8 waves/block share one 16-row A panel in LDS.
template <int KX, bool XF32>
__global__ __launch_bounds__(256) void gru_gates(
    const unsigned short* __restrict__ h_bf, const float* __restrict__ h_f32,
    const unsigned short* __restrict__ xin_bf, const float* __restrict__ xin_f32,
    long xstride,
    const unsigned short* __restrict__ Wz_bf, const float* __restrict__ bz,
    float* __restrict__ z_ws, unsigned short* __restrict__ rh_bf) {
  constexpr int KTOT = DH + KX;
  __shared__ unsigned short sA[16 * (KTOT + 8)];
  const int lane  = threadIdx.x & 31;
  const int wv    = threadIdx.x >> 5;
  const int tm    = blockIdx.x >> 3;                 // 16 row tiles, 8 blocks each
  const int tn    = ((blockIdx.x & 7) << 3) + wv;    // 0..63 over 2H
  const int m0    = tm << 4;
  const int n0    = tn << 4;
  const int am    = lane & 15;
  const int akoff = (lane >> 4) << 3;

  stage_A<KX, XF32>(sA, m0, h_bf, xin_bf, xin_f32, xstride);
  v8f acc = gemm_tile<KTOT>(sA, Wz_bf, 3 * DH, n0, lane, am, akoff);

  // C layout: VGPR r, lanes 0-15 -> M=m0+r, lanes 16-31 -> M=m0+r+8; N=lane&15.
  const int col   = n0 + (lane & 15);                // 0..1023 within [z|r]
  const int rbase = m0 + ((lane >> 4) << 3);
  const float bias = bz[col];
  const bool  is_z = (col < DH);
  const int   hcol = is_z ? col : (col - DH);
#pragma unroll
  for (int r = 0; r < 8; ++r) {
    const int   idx = (rbase + r) * DH + hcol;
    const float g   = fast_sigmoid(acc[r] + bias);
    if (is_z) z_ws[idx] = g;
    else      rh_bf[idx] = f2bf(g * h_f32[idx]);
  }
}

// Candidate+update: h~ = tanh([r*h, x] @ Wc + bc); h = (1-z)h + z h~.
template <int KX, bool XF32>
__global__ __launch_bounds__(256) void gru_cand(
    const unsigned short* __restrict__ rh_bf,
    const unsigned short* __restrict__ xin_bf, const float* __restrict__ xin_f32,
    long xstride,
    const unsigned short* __restrict__ Wc_bf, const float* __restrict__ bc,
    const float* __restrict__ z_ws,
    float* __restrict__ h_f32, unsigned short* __restrict__ h_bf) {
  constexpr int KTOT = DH + KX;
  __shared__ unsigned short sA[16 * (KTOT + 8)];
  const int lane  = threadIdx.x & 31;
  const int wv    = threadIdx.x >> 5;
  const int tm    = blockIdx.x >> 2;                 // 16 row tiles, 4 blocks each
  const int tn    = ((blockIdx.x & 3) << 3) + wv;    // 0..31 over H
  const int m0    = tm << 4;
  const int n0    = tn << 4;
  const int am    = lane & 15;
  const int akoff = (lane >> 4) << 3;

  stage_A<KX, XF32>(sA, m0, rh_bf, xin_bf, xin_f32, xstride);
  v8f acc = gemm_tile<KTOT>(sA, Wc_bf, DH, n0, lane, am, akoff);

  const int col   = n0 + (lane & 15);
  const int rbase = m0 + ((lane >> 4) << 3);
  const float bias = bc[col];
#pragma unroll
  for (int r = 0; r < 8; ++r) {
    const int   idx = (rbase + r) * DH + col;
    const float ht  = fast_tanh(acc[r] + bias);
    const float z   = z_ws[idx];
    const float hn  = (1.0f - z) * h_f32[idx] + z * ht;
    h_f32[idx] = hn;
    h_bf[idx]  = f2bf(hn);
  }
}

__global__ void cvt_f32_bf16(const float* __restrict__ s, unsigned short* __restrict__ d, int n) {
  for (int i = blockIdx.x * blockDim.x + threadIdx.x; i < n; i += gridDim.x * blockDim.x)
    d[i] = f2bf(s[i]);
}
__global__ void zero_f32(float* __restrict__ p, int n) {
  for (int i = blockIdx.x * blockDim.x + threadIdx.x; i < n; i += gridDim.x * blockDim.x)
    p[i] = 0.0f;
}
__global__ void zero_u16(unsigned short* __restrict__ p, int n) {
  for (int i = blockIdx.x * blockDim.x + threadIdx.x; i < n; i += gridDim.x * blockDim.x)
    p[i] = 0;
}
// Output tuple flat: h1 (B*H), then h_n = [h0 (B*H), h1 (B*H)].
__global__ void finalize(const float* __restrict__ h0, const float* __restrict__ h1,
                         float* __restrict__ out) {
  int i = blockIdx.x * blockDim.x + threadIdx.x;
  if (i < B_SZ * DH) {
    out[i] = h1[i];
    out[B_SZ * DH + i] = h0[i];
    out[2 * B_SZ * DH + i] = h1[i];
  }
}

extern "C" void kernel_launch(void* const* d_in, const int* /*in_sizes*/, int /*n_in*/,
                              void* d_out, int /*out_size*/, void* d_ws, size_t /*ws_size*/,
                              hipStream_t stream) {
  const float* x   = (const float*)d_in[0];
  const float* Wz0 = (const float*)d_in[1];
  const float* bz0 = (const float*)d_in[2];
  const float* Wc0 = (const float*)d_in[3];
  const float* bc0 = (const float*)d_in[4];
  const float* Wz1 = (const float*)d_in[5];
  const float* bz1 = (const float*)d_in[6];
  const float* Wc1 = (const float*)d_in[7];
  const float* bc1 = (const float*)d_in[8];

  char* ws = (char*)d_ws;
  size_t off = 0;
  auto take = [&](size_t bytes) -> char* {
    char* p = ws + off;
    off += (bytes + 255) & ~(size_t)255;
    return p;
  };
  unsigned short* wz0b = (unsigned short*)take((size_t)768 * 1536 * 2);
  unsigned short* wc0b = (unsigned short*)take((size_t)768 * 512 * 2);
  unsigned short* wz1b = (unsigned short*)take((size_t)1024 * 1536 * 2);
  unsigned short* wc1b = (unsigned short*)take((size_t)1024 * 512 * 2);
  float*          h0f  = (float*)take((size_t)B_SZ * DH * 4);
  float*          h1f  = (float*)take((size_t)B_SZ * DH * 4);
  unsigned short* h0b  = (unsigned short*)take((size_t)B_SZ * DH * 2);
  unsigned short* h1b  = (unsigned short*)take((size_t)B_SZ * DH * 2);
  float*          zws  = (float*)take((size_t)B_SZ * DH * 4);
  unsigned short* rhb  = (unsigned short*)take((size_t)B_SZ * DH * 2);

  cvt_f32_bf16<<<512, 256, 0, stream>>>(Wz0, wz0b, 768 * 1536);
  cvt_f32_bf16<<<512, 256, 0, stream>>>(Wc0, wc0b, 768 * 512);
  cvt_f32_bf16<<<512, 256, 0, stream>>>(Wz1, wz1b, 1024 * 1536);
  cvt_f32_bf16<<<512, 256, 0, stream>>>(Wc1, wc1b, 1024 * 512);
  zero_f32<<<256, 256, 0, stream>>>(h0f, B_SZ * DH);
  zero_f32<<<256, 256, 0, stream>>>(h1f, B_SZ * DH);
  zero_u16<<<256, 256, 0, stream>>>(h0b, B_SZ * DH);
  zero_u16<<<256, 256, 0, stream>>>(h1b, B_SZ * DH);

  const long xstride = (long)S_SZ * DIN;   // batch stride in x
  for (int t = 0; t < S_SZ; ++t) {
    const float* xt = x + (size_t)t * DIN;
    // layer 0 (x input fp32, K_x = 256)
    gru_gates<DIN, true><<<128, 256, 0, stream>>>(h0b, h0f, nullptr, xt, xstride,
                                                  wz0b, bz0, zws, rhb);
    gru_cand<DIN, true><<<64, 256, 0, stream>>>(rhb, nullptr, xt, xstride,
                                                wc0b, bc0, zws, h0f, h0b);
    // layer 1 (input is layer-0 h bf16, K_x = 512)
    gru_gates<DH, false><<<128, 256, 0, stream>>>(h1b, h1f, h0b, nullptr, 0,
                                                  wz1b, bz1, zws, rhb);
    gru_cand<DH, false><<<64, 256, 0, stream>>>(rhb, h0b, nullptr, 0,
                                                wc1b, bc1, zws, h1f, h1b);
  }

  finalize<<<(B_SZ * DH + 255) / 256, 256, 0, stream>>>(h0f, h1f, (float*)d_out);
}
